// ConvBNN_28724741276341
// MI455X (gfx1250) — 8-bit matrix operand order repaired and checked by reading, compiled, not run on hardware
//
#include <hip/hip_runtime.h>
#include <hip/hip_bf16.h>


typedef int v8i __attribute__((ext_vector_type(8)));

#define KPAD_MAX 768

__device__ __forceinline__ float sgnf(float v) {
    return (v > 0.f) ? 1.f : ((v < 0.f) ? -1.f : 0.f);
}

__global__ __launch_bounds__(256)
void stage1_fused_kernel(const float* __restrict__ x,
                         const float* __restrict__ w1,
                         const float* __restrict__ w2,
                         float* __restrict__ y2, int total) {
    int t = blockIdx.x * blockDim.x + threadIdx.x;
    if (t >= total) return;
    int w  = t % 9;
    int tmp = t / 9;
    int h2 = tmp % 220;
    int nb = tmp / 220;

    float xr[9];
    int base = nb * 1100 * 9 + w;
#pragma unroll
    for (int r = 0; r < 9; ++r) {
        int ih = 5 * h2 - 2 + r;
        xr[r] = (ih >= 0 && ih < 1100) ? x[base + ih * 9] : 0.f;
    }
    float y1loc[16][5];
#pragma unroll
    for (int ic = 0; ic < 16; ++ic) {
#pragma unroll
        for (int k2 = 0; k2 < 5; ++k2) {
            float s = 0.f;
#pragma unroll
            for (int k1 = 0; k1 < 5; ++k1)
                s += sgnf(w1[ic * 5 + k1]) * xr[k2 + k1];
            y1loc[ic][k2] = s;
        }
    }
    int ob = ((nb * 16) * 220 + h2) * 9 + w;
    for (int oc = 0; oc < 16; ++oc) {
        float s = 0.f;
        for (int ic = 0; ic < 16; ++ic)
#pragma unroll
            for (int k2 = 0; k2 < 5; ++k2)
                s += sgnf(w2[(oc * 16 + ic) * 5 + k2]) * y1loc[ic][k2];
        y2[ob + oc * 220 * 9] = s;
    }
}

template <bool SIGN_OUT>
__global__ __launch_bounds__(128)
void binconv_wmma_kernel(const signed char* __restrict__ act,
                         const float* __restrict__ wgt,
                         void* __restrict__ out,
                         int Cin, int Hin, int Win,
                         int Cout, int KH, int KW,
                         int strideH, int strideW, int padH, int padW,
                         int Hout, int Wout, int Kreal, int Kpad, int Mtot) {
    __shared__ __align__(16) signed char sB[16 * KPAD_MAX];
    __shared__ __align__(16) signed char sA[4 * 16 * KPAD_MAX];

    const int tid  = threadIdx.x;
    const int wave = tid >> 5;
    const int lane = tid & 31;
    const int oc0  = blockIdx.y * 16;
    const int tileBase = blockIdx.x * 4;
    const int KHW = KH * KW;
    const int HWo = Hout * Wout;

    for (int idx = tid; idx < 16 * Kpad; idx += 128) {
        int n = idx / Kpad, k = idx - n * Kpad;
        signed char v = 0;
        if (k < Kreal) {
            float wv = wgt[(oc0 + n) * Kreal + k];
            v = (wv > 0.f) ? 1 : ((wv < 0.f) ? -1 : 0);
        }
        sB[n * Kpad + k] = v;
    }
    for (int idx = tid; idx < 4 * 16 * Kpad; idx += 128) {
        int t    = idx / (16 * Kpad);
        int r    = idx - t * 16 * Kpad;
        int mloc = r / Kpad;
        int k    = r - mloc * Kpad;
        int m    = (tileBase + t) * 16 + mloc;
        signed char* dst = &sA[t * 16 * Kpad + mloc * Kpad + k];
        bool valid = false;
        int gidx = 0;
        if (m < Mtot && k < Kreal) {
            int ow  = m % Wout;
            int tm  = m / Wout;
            int oh  = tm % Hout;
            int nb  = tm / Hout;
            int ic  = k / KHW;
            int rr  = k - ic * KHW;
            int kh  = rr / KW;
            int kw  = rr - kh * KW;
            int ih  = oh * strideH + kh - padH;
            int iw  = ow * strideW + kw - padW;
            if ((unsigned)ih < (unsigned)Hin && (unsigned)iw < (unsigned)Win) {
                valid = true;
                gidx = ((nb * Cin + ic) * Hin + ih) * Win + iw;
            }
        }
        if (valid) {
            unsigned ldsaddr = (unsigned)(uintptr_t)dst;
            const signed char* gptr = act + gidx;
            asm volatile("global_load_async_to_lds_b8 %0, %1, off"
                         :
                         : "v"(ldsaddr), "v"(gptr)
                         : "memory");
        } else {
            *dst = 0;
        }
    }
    asm volatile("s_wait_asynccnt 0x0" ::: "memory");
    __syncthreads();

    const int half = lane >> 4;
    const int l15  = lane & 15;
    const int* pA = (const int*)(sA + wave * 16 * Kpad + l15 * Kpad);
    const int* pB = (const int*)(sB + l15 * Kpad);

    v8i acc;
#pragma unroll
    for (int i = 0; i < 8; ++i) acc[i] = 0;

    for (int kc = 0; kc < Kpad; kc += 64) {
        v8i a, b;
#pragma unroll
        for (int i = 0; i < 8; ++i) {
            int ka = kc + ((i >> 2) << 5) + (half << 4) + ((i & 3) << 2);
            a[i] = pA[ka >> 2];
        }
#pragma unroll
        for (int i = 0; i < 8; ++i) {
            int kb = kc + ((i >> 2) << 5) + (half << 4) + ((i & 3) << 2);
            b[i] = pB[kb >> 2];
        }
        acc = __builtin_amdgcn_wmma_i32_16x16x64_iu8(true, a, true, b, acc,
                                                     false, false);
    }

    int tile = tileBase + wave;
    if (tile * 16 < Mtot) {
        int oc = oc0 + l15;
#pragma unroll
        for (int i = 0; i < 8; ++i) {
            int m = tile * 16 + i + half * 8;
            if (m < Mtot) {
                int ow = m % Wout;
                int tm = m / Wout;
                int oh = tm % Hout;
                int nb = tm / Hout;
                int off = (nb * Cout + oc) * HWo + oh * Wout + ow;
                int v = acc[i];
                if (SIGN_OUT)
                    ((signed char*)out)[off] =
                        (v > 0) ? 1 : ((v < 0) ? -1 : 0);
                else
                    ((float*)out)[off] = (float)v;
            }
        }
    }
}

__global__ void zero_kernel(float* __restrict__ p, int n) {
    int i = blockIdx.x * blockDim.x + threadIdx.x;
    if (i < n) p[i] = 0.f;
}

__global__ __launch_bounds__(256)
void stats_kernel(const float* __restrict__ x, float* __restrict__ acc,
                  int total, int C, int HW) {
    __shared__ float ls[256];
    __shared__ float lq[256];
    for (int c = threadIdx.x; c < C; c += blockDim.x) { ls[c] = 0.f; lq[c] = 0.f; }
    __syncthreads();
    for (int i = blockIdx.x * blockDim.x + threadIdx.x; i < total;
         i += gridDim.x * blockDim.x) {
        float v = x[i];
        int c = (i / HW) % C;
        atomicAdd(&ls[c], v);
        atomicAdd(&lq[c], v * v);
    }
    __syncthreads();
    for (int c = threadIdx.x; c < C; c += blockDim.x) {
        atomicAdd(&acc[c], ls[c]);
        atomicAdd(&acc[C + c], lq[c]);
    }
}

__global__ void bnfin_kernel(const float* __restrict__ acc,
                             const float* __restrict__ g,
                             const float* __restrict__ b,
                             float* __restrict__ scale,
                             float* __restrict__ shift, int C, float invCnt) {
    int c = blockIdx.x * blockDim.x + threadIdx.x;
    if (c >= C) return;
    float mean = acc[c] * invCnt;
    float var  = acc[C + c] * invCnt - mean * mean;
    float sc   = g[c] * rsqrtf(var + 1e-5f);
    scale[c] = sc;
    shift[c] = b[c] - mean * sc;
}

template <int SIGN>
__global__ __launch_bounds__(256)
void bn_prelu_kernel(const float* __restrict__ x,
                     const float* __restrict__ scale,
                     const float* __restrict__ shift,
                     const float* __restrict__ aP, void* __restrict__ out,
                     int total, int C, int HW) {
    int i = blockIdx.x * blockDim.x + threadIdx.x;
    if (i >= total) return;
    int c = (i / HW) % C;
    float v = x[i] * scale[c] + shift[c];
    float a = aP[0];
    v = (v >= 0.f) ? v : a * v;
    if (SIGN)
        ((signed char*)out)[i] = (v > 0.f) ? 1 : ((v < 0.f) ? -1 : 0);
    else
        ((float*)out)[i] = v;
}

__global__ __launch_bounds__(256)
void fc1_kernel(const float* __restrict__ h, const float* __restrict__ fw,
                float* __restrict__ out, int Nb, int Cout, int K) {
    int t = blockIdx.x * blockDim.x + threadIdx.x;
    if (t >= Nb * Cout) return;
    int oc = t % Cout;
    int nb = t / Cout;
    const float* hp = h + nb * K;
    const float* wp = fw + oc * K;
    float s = 0.f;
    for (int k = 0; k < K; ++k) {
        float wv = wp[k];
        s += (wv > 0.f) ? hp[k] : ((wv < 0.f) ? -hp[k] : 0.f);
    }
    out[nb * Cout + oc] = s;
}

__global__ __launch_bounds__(256)
void fc3_kernel(const signed char* __restrict__ a, const float* __restrict__ fw,
                float* __restrict__ out, int Nb, int K) {
    int t = blockIdx.x * blockDim.x + threadIdx.x;
    if (t >= Nb * 2) return;
    int j  = t & 1;
    int nb = t >> 1;
    int s = 0;
    for (int k = 0; k < K; ++k) {
        float wv = fw[j * K + k];
        int sw = (wv > 0.f) ? 1 : ((wv < 0.f) ? -1 : 0);
        s += (int)a[nb * K + k] * sw;
    }
    out[nb * 2 + j] = (float)s;
}

extern "C" void kernel_launch(void* const* d_in, const int* in_sizes, int n_in,
                              void* d_out, int out_size, void* d_ws,
                              size_t ws_size, hipStream_t stream) {
    (void)in_sizes; (void)n_in; (void)out_size; (void)ws_size;
    const float* x   = (const float*)d_in[0];
    const float* w[11];
    for (int i = 1; i <= 10; ++i) w[i] = (const float*)d_in[i];
    const float* gP[8], *bP[8], *aP[8];
    for (int s = 0; s < 5; ++s) {
        gP[s + 1] = (const float*)d_in[11 + 3 * s];
        bP[s + 1] = (const float*)d_in[12 + 3 * s];
        aP[s + 1] = (const float*)d_in[13 + 3 * s];
    }
    const float* fc1w = (const float*)d_in[26];
    const float* fc2w = (const float*)d_in[27];
    const float* fc3w = (const float*)d_in[28];
    gP[6] = (const float*)d_in[29]; bP[6] = (const float*)d_in[30];
    gP[7] = (const float*)d_in[31]; bP[7] = (const float*)d_in[32];
    aP[6] = (const float*)d_in[33]; aP[7] = (const float*)d_in[34];

    char* ws = (char*)d_ws;
    size_t o = 0;
    auto carve = [&](size_t bytes) {
        size_t r = o;
        o += (bytes + 255) & ~(size_t)255;
        return r;
    };
    float*       bufY   = (float*)(ws + carve((size_t)16220160 * 4));
    signed char* bufA   = (signed char*)(ws + carve(16220160));
    signed char* bufT   = (signed char*)(ws + carve(32440320));
    float*       stats  = (float*)(ws + carve(512 * 4));
    float*       scaleB = (float*)(ws + carve(256 * 4));
    float*       shiftB = (float*)(ws + carve(256 * 4));

    auto run_binconv = [&](const signed char* actIn, const float* wgt,
                           void* outP, bool signOut, int Cin, int Hin, int Win,
                           int Cout, int KH, int KW, int sH, int sW, int pH,
                           int pW) {
        int Hout  = (Hin + 2 * pH - KH) / sH + 1;
        int Wout  = (Win + 2 * pW - KW) / sW + 1;
        int Kreal = Cin * KH * KW;
        int Kpad  = (Kreal + 63) & ~63;
        int Mtot  = 512 * Hout * Wout;
        int mtiles = (Mtot + 15) / 16;
        dim3 grid((mtiles + 3) / 4, Cout / 16);
        if (signOut)
            binconv_wmma_kernel<true><<<grid, 128, 0, stream>>>(
                actIn, wgt, outP, Cin, Hin, Win, Cout, KH, KW, sH, sW, pH, pW,
                Hout, Wout, Kreal, Kpad, Mtot);
        else
            binconv_wmma_kernel<false><<<grid, 128, 0, stream>>>(
                actIn, wgt, outP, Cin, Hin, Win, Cout, KH, KW, sH, sW, pH, pW,
                Hout, Wout, Kreal, Kpad, Mtot);
    };

    auto run_bn = [&](const float* y, const float* g, const float* b,
                      const float* a, void* outP, int sign, int total, int C,
                      int HW) {
        zero_kernel<<<2, 256, 0, stream>>>(stats, 2 * C);
        int blocks = (total + 255) / 256;
        if (blocks > 4096) blocks = 4096;
        stats_kernel<<<blocks, 256, 0, stream>>>(y, stats, total, C, HW);
        bnfin_kernel<<<1, 256, 0, stream>>>(stats, g, b, scaleB, shiftB, C,
                                            1.0f / (float)(total / C));
        int ablocks = (total + 255) / 256;
        if (sign)
            bn_prelu_kernel<1><<<ablocks, 256, 0, stream>>>(
                y, scaleB, shiftB, a, outP, total, C, HW);
        else
            bn_prelu_kernel<0><<<ablocks, 256, 0, stream>>>(
                y, scaleB, shiftB, a, outP, total, C, HW);
    };

    {
        int total1 = 512 * 220 * 9;
        stage1_fused_kernel<<<(total1 + 255) / 256, 256, 0, stream>>>(
            x, w[1], w[2], bufY, total1);
        run_bn(bufY, gP[1], bP[1], aP[1], bufA, 1, 512 * 16 * 220 * 9, 16,
               220 * 9);
    }
    run_binconv(bufA, w[3], bufT, true, 16, 220, 9, 32, 5, 1, 1, 1, 2, 0);
    run_binconv(bufT, w[4], bufY, false, 32, 220, 9, 32, 5, 1, 5, 1, 0, 0);
    run_bn(bufY, gP[2], bP[2], aP[2], bufA, 1, 512 * 32 * 44 * 9, 32, 44 * 9);
    run_binconv(bufA, w[5], bufT, true, 32, 44, 9, 64, 11, 1, 1, 1, 5, 0);
    run_binconv(bufT, w[6], bufY, false, 64, 44, 9, 64, 11, 1, 11, 1, 0, 0);
    run_bn(bufY, gP[3], bP[3], aP[3], bufA, 1, 512 * 64 * 4 * 9, 64, 36);
    run_binconv(bufA, w[7], bufT, true, 64, 4, 9, 128, 1, 3, 1, 1, 0, 1);
    run_binconv(bufT, w[8], bufY, false, 128, 4, 9, 128, 1, 3, 1, 3, 0, 0);
    run_bn(bufY, gP[4], bP[4], aP[4], bufA, 1, 512 * 128 * 4 * 3, 128, 12);
    run_binconv(bufA, w[9], bufT, true, 128, 4, 3, 256, 1, 3, 1, 1, 0, 1);
    run_binconv(bufT, w[10], bufY, false, 256, 4, 3, 256, 1, 3, 1, 3, 0, 0);
    run_bn(bufY, gP[5], bP[5], aP[5], bufA, 0, 512 * 256 * 4 * 1, 256, 4);

    fc1_kernel<<<(512 * 256 + 255) / 256, 256, 0, stream>>>(
        (const float*)bufA, fc1w, bufY, 512, 256, 1024);
    run_bn(bufY, gP[6], bP[6], aP[6], bufA, 1, 512 * 256, 256, 1);
    run_binconv(bufA, fc2w, bufY, false, 256, 1, 1, 64, 1, 1, 1, 1, 0, 0);
    run_bn(bufY, gP[7], bP[7], aP[7], bufA, 1, 512 * 64, 64, 1);
    fc3_kernel<<<(512 * 2 + 255) / 256, 256, 0, stream>>>(
        (const signed char*)bufA, fc3w, (float*)d_out, 512, 64);
}
